// Model_73701638800049
// MI455X (gfx1250) — matrix operand order repaired and checked by reading, not compiled, not run on hardware
//
#include <hip/hip_runtime.h>
#include <hip/hip_bf16.h>

// ---------------------------------------------------------------- types
typedef __attribute__((ext_vector_type(16))) __bf16 v16bf;
typedef __attribute__((ext_vector_type(8)))  float  v8f;

union FragBF { v16bf v; unsigned short u[16]; unsigned int d[8]; uint4 q[2]; };

__device__ __forceinline__ unsigned short f2bf(float x) {
  unsigned int u = __float_as_uint(x);
  unsigned int r = u + 0x7fffu + ((u >> 16) & 1u);
  return (unsigned short)(r >> 16);
}

// ---------------------------------------------------------------- WMMA GEMM
// C[b](MxN,f32) = A[b](MxK bf16 row-major) * Bp[b](KxN bf16, PACKED)
// Packed B layout: Bp[((nt*(K/32)+kb)*32+lane)*16+s] = B[kb*32+16*(lane/16)+s][nt*16+lane%16]
// Each wave computes a 16x32 tile (two WMMAs share one A fragment).
// EPI bits: 1 = bias[col], 2 = rowscale[b*M+row], 4 = relu
template <int EPI>
__global__ void wmma_gemm_bf16(const unsigned short* __restrict__ A,
                               const unsigned short* __restrict__ B,
                               float* __restrict__ C,
                               int M, int N, int K,
                               long long sA, long long sB, long long sC,
                               const float* __restrict__ bias,
                               const float* __restrict__ rowscale) {
  int bz = blockIdx.z;
  A += bz * sA; B += bz * sB; C += bz * sC;
  int wave = threadIdx.x >> 5;
  int lane = threadIdx.x & 31;
  int m0 = (blockIdx.y * 4 + wave) * 16;
  int n0 = blockIdx.x * 32;                     // two 16-wide n-tiles
  if (m0 >= M) return;
  int half = lane >> 4;
  int l16  = lane & 15;
  int arow = m0 + l16;
  int kb5  = K >> 5;                            // K/32 blocks
  const unsigned short* arp = A + (long long)arow * K;
  // per-lane packed-B base for tile nt and k-block kb: ((nt*kb5+kb)*32+lane)*16
  long long bbase0 = (((long long)(n0 >> 4) * kb5) * 32 + lane) * 16;
  long long bbase1 = bbase0 + (long long)kb5 * 512;   // next n-tile
  v8f acc0 = {}, acc1 = {};
#pragma unroll 2
  for (int kk = 0; kk < K; kk += 32) {
    FragBF fa, fb0, fb1;
    fa.q[0] = *(const uint4*)(arp + kk + (half << 4));
    fa.q[1] = *(const uint4*)(arp + kk + 8 + (half << 4));
    const unsigned short* bp0 = B + bbase0 + ((long long)(kk >> 5) << 9);
    const unsigned short* bp1 = B + bbase1 + ((long long)(kk >> 5) << 9);
    fb0.q[0] = *(const uint4*)(bp0);
    fb0.q[1] = *(const uint4*)(bp0 + 8);
    fb1.q[0] = *(const uint4*)(bp1);
    fb1.q[1] = *(const uint4*)(bp1 + 8);
    acc0 = __builtin_amdgcn_wmma_f32_16x16x32_bf16(
        false, fa.v, false, fb0.v, (short)0, acc0, false, false);
    acc1 = __builtin_amdgcn_wmma_f32_16x16x32_bf16(
        false, fa.v, false, fb1.v, (short)0, acc1, false, false);
  }
  float bias0 = 0.0f, bias1 = 0.0f;
  if (EPI & 1) { bias0 = bias[n0 + l16]; bias1 = bias[n0 + 16 + l16]; }
#pragma unroll
  for (int r = 0; r < 8; ++r) {
    int row = m0 + r + (half << 3);             // C/D vgpr r -> M = r (+8 hi half)
    float v0 = acc0[r], v1 = acc1[r];
    if (EPI & 2) {
      float rs = rowscale[(long long)bz * M + row];
      v0 *= rs; v1 *= rs;
    }
    if (EPI & 1) { v0 += bias0; v1 += bias1; }
    if (EPI & 4) { v0 = fmaxf(v0, 0.0f); v1 = fmaxf(v1, 0.0f); }
    C[(long long)row * N + n0 + l16]      = v0;
    C[(long long)row * N + n0 + 16 + l16] = v1;
  }
}

// ---------------------------------------------------------------- conversions
__global__ void k_f32_to_bf16(const float* __restrict__ in,
                              unsigned short* __restrict__ out, long long n) {
  long long i = (long long)blockIdx.x * blockDim.x + threadIdx.x;
  if (i < n) out[i] = f2bf(in[i]);
}
// f32 [.., K, N] row-major -> packed bf16 B layout (see wmma_gemm_bf16)
__global__ void k_pack_bf16(const float* __restrict__ src,
                            unsigned short* __restrict__ out,
                            int K, int N, long long total) {
  long long id = (long long)blockIdx.x * blockDim.x + threadIdx.x;
  if (id >= total) return;
  long long m = (long long)K * N;
  long long b = id / m;
  long long r = id % m;
  int s    = (int)(r & 15);
  int lane = (int)((r >> 4) & 31);
  long long t = r >> 9;
  int kb5 = K >> 5;
  int kb = (int)(t % kb5);
  int nt = (int)(t / kb5);
  int k = (kb << 5) + ((lane >> 4) << 4) + s;
  int n = (nt << 4) + (lane & 15);
  out[id] = f2bf(src[b * m + (long long)k * N + n]);
}
__global__ void k_zero_f32(float* p, long long n) {
  long long i = (long long)blockIdx.x * blockDim.x + threadIdx.x;
  if (i < n) p[i] = 0.0f;
}
__global__ void k_fill_int(int* p, int v, long long n) {
  long long i = (long long)blockIdx.x * blockDim.x + threadIdx.x;
  if (i < n) p[i] = v;
}

// ---------------------------------------------------------------- graph (sparse)
__global__ void k_deg(const int* __restrict__ dst, float* __restrict__ deg, int E) {
  int e = blockIdx.x * blockDim.x + threadIdx.x;
  if (e < E) atomicAdd(&deg[dst[e]], 1.0f);
}
__global__ void k_dinv(const float* deg, float* dinv, long long n) {
  long long i = (long long)blockIdx.x * blockDim.x + threadIdx.x;
  if (i < n) dinv[i] = rsqrtf(deg[i] + 1.0f);   // self-loop included, >=1
}
__global__ void k_selfinit(const float* __restrict__ xw, const float* __restrict__ dinv,
                           float* __restrict__ h, long long n) {
  long long i = (long long)blockIdx.x * blockDim.x + threadIdx.x;
  if (i < n) { long long r = i >> 8; float d = dinv[r]; h[i] = xw[i] * d * d; }
}
// one wave per edge: 32 lanes x 8 floats = one 256-feature row
__global__ void k_scatter(const int* __restrict__ src, const int* __restrict__ dst,
                          const float* __restrict__ xin, const float* __restrict__ dinv,
                          float* __restrict__ out, int E) {
  long long gid = (long long)blockIdx.x * blockDim.x + threadIdx.x;
  int e = (int)(gid >> 5);
  if (e >= E) return;
  int base = ((int)gid & 31) * 8;
  int s = src[e], d = dst[e];
  float c = dinv ? dinv[s] * dinv[d] : 1.0f;
  const float* xp = xin + (long long)s * 256 + base;
  float* op = out + (long long)d * 256 + base;
#pragma unroll
  for (int j = 0; j < 8; ++j) atomicAdd(op + j, xp[j] * c);
}
__global__ void k_bias_relu(float* __restrict__ h, const float* __restrict__ b, long long n) {
  long long i = (long long)blockIdx.x * blockDim.x + threadIdx.x;
  if (i < n) h[i] = fmaxf(h[i] + b[i & 255], 0.0f);
}

// ---------------------------------------------------------------- row reductions (wave/row)
__global__ void k_score(const float* __restrict__ h, const float* __restrict__ nb,
                        const float* __restrict__ deg, float* __restrict__ sc,
                        int rows, int feat) {
  int wid = (blockIdx.x * blockDim.x + threadIdx.x) >> 5;
  int lane = threadIdx.x & 31;
  if (wid >= rows) return;
  float inv = deg ? 1.0f / fmaxf(deg[wid], 1.0f) : 1.0f;
  const float* hr = h + (long long)wid * feat;
  const float* nr = nb + (long long)wid * feat;
  float s = 0.0f;
  for (int j = lane; j < feat; j += 32) s += fabsf(hr[j] - nr[j] * inv);
  for (int o = 16; o; o >>= 1) s += __shfl_xor(s, o);
  if (!lane) sc[wid] = s;
}
__global__ void k_rowstat(const float* __restrict__ adj, float* __restrict__ out,
                          int rows, int k, int mode) {
  int wid = (blockIdx.x * blockDim.x + threadIdx.x) >> 5;
  int lane = threadIdx.x & 31;
  if (wid >= rows) return;
  const float* r = adj + (long long)wid * k;
  float s = 0.0f;
  for (int j = lane; j < k; j += 32) s += r[j];
  for (int o = 16; o; o >>= 1) s += __shfl_xor(s, o);
  if (!lane) out[wid] = mode ? (1.0f / fmaxf(s, 1e-9f)) : rsqrtf(s + 1.0f);
}
__global__ void k_ef(const float* __restrict__ xsel, const float* __restrict__ att,
                     float* __restrict__ e, float* __restrict__ f, int rows, int feat) {
  int wid = (blockIdx.x * blockDim.x + threadIdx.x) >> 5;
  int lane = threadIdx.x & 31;
  if (wid >= rows) return;
  const float* xr = xsel + (long long)wid * feat;
  float a0 = 0.0f, a1 = 0.0f;
  for (int j = lane; j < feat; j += 32) {
    float v = xr[j];
    a0 += v * att[2 * j];
    a1 += v * att[2 * j + 1];
  }
  for (int o = 16; o; o >>= 1) { a0 += __shfl_xor(a0, o); a1 += __shfl_xor(a1, o); }
  if (!lane) { e[wid] = a0; f[wid] = a1; }
}

// ---------------------------------------------------------------- top-k / gather
// stable descending selection == lax.top_k semantics
__global__ void k_topk(const float* __restrict__ score, int* __restrict__ idx,
                       int nin, int k) {
  int b = blockIdx.x, t = threadIdx.x;
  __shared__ float sv[512];
  float s = score[(long long)b * nin + t];
  sv[t] = s;
  __syncthreads();
  int r = 0;
  for (int j = 0; j < nin; ++j) {
    float sj = sv[j];
    r += (sj > s) || (sj == s && j < t);
  }
  if (r < k) idx[b * k + r] = t;
}
__global__ void k_rank(const int* __restrict__ idx, int* __restrict__ rank,
                       int B, int k, int n) {
  int i = blockIdx.x * blockDim.x + threadIdx.x;
  if (i >= B * k) return;
  int b = i / k, r = i % k;
  rank[b * n + idx[i]] = r;
}
__global__ void k_gather_rows(const float* __restrict__ src, const int* __restrict__ idx,
                              float* __restrict__ dst, int B, int k, int feat, int srcRows) {
  long long id = (long long)blockIdx.x * blockDim.x + threadIdx.x;
  long long n = (long long)B * k * feat;
  if (id >= n) return;
  int f = (int)(id % feat);
  long long t = id / feat;
  int r = (int)(t % k);
  int b = (int)(t / k);
  int s = idx[b * k + r];
  dst[id] = src[((long long)b * srcRows + s) * feat + f];
}
__global__ void k_gather_adj(const float* __restrict__ adj, const int* __restrict__ idx,
                             float* __restrict__ out, int B, int kin, int kout) {
  long long id = (long long)blockIdx.x * blockDim.x + threadIdx.x;
  long long n = (long long)B * kout * kout;
  if (id >= n) return;
  int j = (int)(id % kout);
  long long t = id / kout;
  int i = (int)(t % kout);
  int b = (int)(t / kout);
  out[id] = adj[((long long)b * kin + idx[b * kout + i]) * kin + idx[b * kout + j]];
}
__global__ void k_adj_build(const int* __restrict__ src, const int* __restrict__ dst,
                            const int* __restrict__ rank, float* __restrict__ adj,
                            int E, int k) {
  int e = blockIdx.x * blockDim.x + threadIdx.x;
  if (e >= E) return;
  int s = src[e], d = dst[e];
  int g = s >> 9;                       // N = 512
  int ns = rank[s], nd = rank[d];
  if (ns >= 0 && nd >= 0) adj[((long long)g * k + ns) * k + nd] = 1.0f;
}

// ---------------------------------------------------------------- softmax / misc
__global__ void k_softmax_adj(float* __restrict__ adj, const float* __restrict__ e,
                              const float* __restrict__ f, int k) {
  int row = blockIdx.x;                 // row = b*k + i
  int b = row / k, t = threadIdx.x;
  __shared__ float red[256];
  float* ar = adj + (long long)row * k;
  float val = e[row] + f[b * k + t] + ar[t];   // LAMB = 1.0
  red[t] = val; __syncthreads();
  for (int s = k >> 1; s > 0; s >>= 1) { if (t < s) red[t] = fmaxf(red[t], red[t + s]); __syncthreads(); }
  float m = red[0]; __syncthreads();
  float ex = __expf(val - m);
  red[t] = ex; __syncthreads();
  for (int s = k >> 1; s > 0; s >>= 1) { if (t < s) red[t] += red[t + s]; __syncthreads(); }
  ar[t] = ex / red[0];
}
__global__ void k_readout(const float* __restrict__ x, float* __restrict__ out,
                          int B, int k, int feat) {
  int id = blockIdx.x * blockDim.x + threadIdx.x;
  if (id >= B * feat) return;
  int b = id / feat, f = id % feat;
  const float* p = x + ((long long)b * k) * feat + f;
  float mx = -INFINITY, sm = 0.0f;
  for (int i = 0; i < k; ++i) { float v = p[(long long)i * feat]; mx = fmaxf(mx, v); sm += v; }
  out[(long long)b * (2 * feat) + f] = mx;
  out[(long long)b * (2 * feat) + feat + f] = sm / (float)k;
}
__global__ void k_bn_bf16(const float* __restrict__ h, unsigned short* __restrict__ out,
                          const float* g, const float* be, const float* mu,
                          const float* var, long long n) {
  long long i = (long long)blockIdx.x * blockDim.x + threadIdx.x;
  if (i >= n) return;
  int f = (int)(i & 255);
  float v = (h[i] - mu[f]) * rsqrtf(var[f] + 1e-5f) * g[f] + be[f];
  out[i] = f2bf(v);
}
__global__ void k_make_anbf(const float* __restrict__ adj, const float* __restrict__ dinv,
                            unsigned short* __restrict__ out, long long n, int k) {
  long long id = (long long)blockIdx.x * blockDim.x + threadIdx.x;
  if (id >= n) return;
  int j = (int)(id % k);
  long long t = id / k;
  int i = (int)(t % k);
  int b = (int)(t / k);
  float v = adj[id] + (i == j ? 1.0f : 0.0f);
  out[id] = f2bf(v * dinv[b * k + i] * dinv[b * k + j]);
}
__global__ void k_relu_sum3(const float* a, const float* b, const float* c,
                            float* o, long long n) {
  long long i = (long long)blockIdx.x * blockDim.x + threadIdx.x;
  if (i < n) o[i] = fmaxf(a[i], 0.0f) + fmaxf(b[i], 0.0f) + fmaxf(c[i], 0.0f);
}
__global__ void k_head_gemm(const float* __restrict__ X, const float* __restrict__ W,
                            const float* __restrict__ bias, float* __restrict__ out,
                            int B, int Kin, int Nout, int relu) {
  int id = blockIdx.x * blockDim.x + threadIdx.x;
  if (id >= B * Nout) return;
  int b = id / Nout, j = id % Nout;
  float a = bias ? bias[j] : 0.0f;
  for (int i = 0; i < Kin; ++i) a += X[b * Kin + i] * W[i * Nout + j];
  if (relu) a = fmaxf(a, 0.0f);
  out[id] = a;
}
__global__ void k_logsoftmax(const float* __restrict__ lg, float* __restrict__ out,
                             int B, int C) {
  int b = blockIdx.x * blockDim.x + threadIdx.x;
  if (b >= B) return;
  const float* r = lg + b * C;
  float m = -INFINITY;
  for (int c = 0; c < C; ++c) m = fmaxf(m, r[c]);
  float s = 0.0f;
  for (int c = 0; c < C; ++c) s += __expf(r[c] - m);
  float ls = logf(s);
  for (int c = 0; c < C; ++c) out[b * C + c] = r[c] - m - ls;
}

// ---------------------------------------------------------------- launch
extern "C" void kernel_launch(void* const* d_in, const int* in_sizes, int n_in,
                              void* d_out, int out_size, void* d_ws, size_t ws_size,
                              hipStream_t stream) {
  (void)in_sizes; (void)n_in; (void)out_size; (void)ws_size;
  constexpr long long NN = 65536;      // B*N nodes
  constexpr int EDGES = 1048576;       // B*EPG
  constexpr int Bb = 128;

  const float* x    = (const float*)d_in[0];
  const int*   ei   = (const int*)d_in[1];
  const float* W1   = (const float*)d_in[3];
  const float* b1   = (const float*)d_in[4];
  const float* W2   = (const float*)d_in[5];
  const float* b2   = (const float*)d_in[6];
  const float* W3   = (const float*)d_in[7];
  const float* b3   = (const float*)d_in[8];
  const float* att1 = (const float*)d_in[9];
  const float* att2 = (const float*)d_in[10];
  const float* bng  = (const float*)d_in[11];
  const float* bnb  = (const float*)d_in[12];
  const float* bnm  = (const float*)d_in[13];
  const float* bnv  = (const float*)d_in[14];
  const float* l1w  = (const float*)d_in[15];
  const float* l1b  = (const float*)d_in[16];
  const float* l2w  = (const float*)d_in[17];
  const float* l2b  = (const float*)d_in[18];
  const float* l3w  = (const float*)d_in[19];
  const float* l3b  = (const float*)d_in[20];
  const int* srcp = ei;
  const int* dstp = ei + EDGES;

  float* out  = (float*)d_out;
  float* xout = out;                   // x_  [128,128]
  float* logp = out + 16384;           // logp [128,10]
  float* hbuf = out + 17664;           // x__ [65536,256]

  float* ws = (float*)d_ws;
  size_t o = 0;
  auto take = [&](size_t n) { size_t r = o; o += n; return r; };
  size_t slabA = take(16777216);       // 64MB multi-use
  size_t slabB = take(16777216);       // 64MB multi-use
  size_t w1bf  = take(65536);
  size_t w2bf  = take(32768);
  size_t w3bf  = take(32768);
  size_t deg1  = take(65536);
  size_t dinv1 = take(65536);
  size_t sc1   = take(65536);
  size_t rankb = take(65536);          // int
  size_t idx1  = take(32768);          // int
  size_t idx2  = take(16384);          // int
  size_t h1    = take(8388608);
  size_t adj1  = take(8388608);
  size_t adj2  = take(2097152);
  size_t e1 = take(32768), f1o = take(32768);
  size_t e2 = take(16384), f2o = take(16384);
  size_t x1 = take(65536), x2 = take(65536), x3 = take(65536);
  size_t dinvA = take(32768);
  size_t sc2   = take(32768);
  size_t gsum  = take(65536);
  size_t g1    = take(32768);
  size_t lgts  = take(1280);

  unsigned short* xbf_p   = (unsigned short*)(ws + slabA);
  float*          xw1_p   = ws + slabB;
  float*          neigh_p = ws + slabA;                               // reuse
  unsigned short* hbbf_p  = (unsigned short*)(ws + slabB);            // stage4
  float*          xw2_p   = ws + slabB + 4194304;
  unsigned short* xw2bf_p = (unsigned short*)(ws + slabB + 12582912);
  unsigned short* anbf1_p = (unsigned short*)(ws + slabA);
  float*          h2_p    = ws + slabA + 4194304;
  unsigned short* h2bf_p  = (unsigned short*)(ws + slabB);            // stage5
  float*          prop_p  = ws + slabB + 4194304;
  unsigned short* a1bf_p  = (unsigned short*)(ws + slabB + 12582912);
  float*          h2sel_p = ws + slabA + 12582912;
  unsigned short* anbf2_p = (unsigned short*)(ws + slabB);            // stage7
  unsigned short* hslbf_p = (unsigned short*)(ws + slabB + 1048576);
  float*          xw3_p   = ws + slabB + 3145728;
  unsigned short* xw3bf_p = (unsigned short*)(ws + slabB + 7340032);
  float*          h3_p    = ws + slabA;

  // ---- stage 1: h = relu(GCN_sparse(x, W1, b1))  (h -> x__ region)
  k_f32_to_bf16<<<131072, 256, 0, stream>>>(x, xbf_p, NN * 512);
  k_pack_bf16<<<512, 256, 0, stream>>>(W1, (unsigned short*)(ws + w1bf), 512, 256, 131072);
  k_pack_bf16<<<256, 256, 0, stream>>>(W2, (unsigned short*)(ws + w2bf), 256, 256, 65536);
  k_pack_bf16<<<256, 256, 0, stream>>>(W3, (unsigned short*)(ws + w3bf), 256, 256, 65536);
  wmma_gemm_bf16<0><<<dim3(8, 1024, 1), 128, 0, stream>>>(
      xbf_p, (unsigned short*)(ws + w1bf), xw1_p, 65536, 256, 512, 0, 0, 0,
      nullptr, nullptr);
  k_zero_f32<<<256, 256, 0, stream>>>(ws + deg1, 65536);
  k_deg<<<4096, 256, 0, stream>>>(dstp, ws + deg1, EDGES);
  k_dinv<<<256, 256, 0, stream>>>(ws + deg1, ws + dinv1, 65536);
  k_selfinit<<<65536, 256, 0, stream>>>(xw1_p, ws + dinv1, hbuf, NN * 256);
  k_scatter<<<131072, 256, 0, stream>>>(srcp, dstp, xw1_p, ws + dinv1, hbuf, EDGES);
  k_bias_relu<<<65536, 256, 0, stream>>>(hbuf, b1, NN * 256);

  // ---- stage 2: pool_sparse -> h1, adj1
  k_zero_f32<<<65536, 256, 0, stream>>>(neigh_p, NN * 256);
  k_scatter<<<131072, 256, 0, stream>>>(srcp, dstp, hbuf, nullptr, neigh_p, EDGES);
  k_score<<<8192, 256, 0, stream>>>(hbuf, neigh_p, ws + deg1, ws + sc1, 65536, 256);
  k_topk<<<128, 512, 0, stream>>>(ws + sc1, (int*)(ws + idx1), 512, 256);
  k_fill_int<<<256, 256, 0, stream>>>((int*)(ws + rankb), -1, 65536);
  k_rank<<<128, 256, 0, stream>>>((int*)(ws + idx1), (int*)(ws + rankb), Bb, 256, 512);
  k_gather_rows<<<32768, 256, 0, stream>>>(hbuf, (int*)(ws + idx1), ws + h1, Bb, 256, 256, 512);
  k_zero_f32<<<32768, 256, 0, stream>>>(ws + adj1, 8388608);
  k_adj_build<<<4096, 256, 0, stream>>>(srcp, dstp, (int*)(ws + rankb), ws + adj1, EDGES, 256);
  k_ef<<<4096, 256, 0, stream>>>(ws + h1, att1, ws + e1, ws + f1o, 32768, 256);
  k_softmax_adj<<<32768, 256, 0, stream>>>(ws + adj1, ws + e1, ws + f1o, 256);
  k_readout<<<128, 256, 0, stream>>>(ws + h1, ws + x1, Bb, 256, 256);

  // ---- stage 4: h2 = relu(gcn_dense(BN(h1), adj1, W2, b2))
  k_bn_bf16<<<32768, 256, 0, stream>>>(ws + h1, hbbf_p, bng, bnb, bnm, bnv, 8388608);
  k_rowstat<<<4096, 256, 0, stream>>>(ws + adj1, ws + dinvA, 32768, 256, 0);
  k_make_anbf<<<32768, 256, 0, stream>>>(ws + adj1, ws + dinvA, anbf1_p, 8388608, 256);
  wmma_gemm_bf16<0><<<dim3(8, 512, 1), 128, 0, stream>>>(
      hbbf_p, (unsigned short*)(ws + w2bf), xw2_p, 32768, 256, 256, 0, 0, 0,
      nullptr, nullptr);
  k_pack_bf16<<<32768, 256, 0, stream>>>(xw2_p, xw2bf_p, 256, 256, 8388608);
  wmma_gemm_bf16<5><<<dim3(8, 4, 128), 128, 0, stream>>>(
      anbf1_p, xw2bf_p, h2_p, 256, 256, 256, 65536, 65536, 65536, b2, nullptr);

  // ---- stage 5: pool_dense -> h2sel, adj2
  k_pack_bf16<<<32768, 256, 0, stream>>>(h2_p, h2bf_p, 256, 256, 8388608);
  k_rowstat<<<4096, 256, 0, stream>>>(ws + adj1, ws + dinvA, 32768, 256, 1);
  k_f32_to_bf16<<<32768, 256, 0, stream>>>(ws + adj1, a1bf_p, 8388608);
  wmma_gemm_bf16<2><<<dim3(8, 4, 128), 128, 0, stream>>>(
      a1bf_p, h2bf_p, prop_p, 256, 256, 256, 65536, 65536, 65536,
      nullptr, ws + dinvA);
  k_score<<<4096, 256, 0, stream>>>(h2_p, prop_p, nullptr, ws + sc2, 32768, 256);
  k_topk<<<128, 256, 0, stream>>>(ws + sc2, (int*)(ws + idx2), 256, 128);
  k_gather_rows<<<16384, 256, 0, stream>>>(h2_p, (int*)(ws + idx2), h2sel_p, Bb, 128, 256, 256);
  k_gather_adj<<<8192, 256, 0, stream>>>(ws + adj1, (int*)(ws + idx2), ws + adj2, Bb, 256, 128);
  k_ef<<<2048, 256, 0, stream>>>(h2sel_p, att2, ws + e2, ws + f2o, 16384, 256);
  k_softmax_adj<<<16384, 128, 0, stream>>>(ws + adj2, ws + e2, ws + f2o, 128);
  k_readout<<<128, 256, 0, stream>>>(h2sel_p, ws + x2, Bb, 128, 256);

  // ---- stage 7: h3 = relu(gcn_dense(h2sel, adj2, W3, b3))
  k_rowstat<<<2048, 256, 0, stream>>>(ws + adj2, ws + dinvA, 16384, 128, 0);
  k_make_anbf<<<8192, 256, 0, stream>>>(ws + adj2, ws + dinvA, anbf2_p, 2097152, 128);
  k_f32_to_bf16<<<16384, 256, 0, stream>>>(h2sel_p, hslbf_p, 4194304);
  wmma_gemm_bf16<0><<<dim3(8, 256, 1), 128, 0, stream>>>(
      hslbf_p, (unsigned short*)(ws + w3bf), xw3_p, 16384, 256, 256, 0, 0, 0,
      nullptr, nullptr);
  k_pack_bf16<<<16384, 256, 0, stream>>>(xw3_p, xw3bf_p, 128, 256, 4194304);
  wmma_gemm_bf16<5><<<dim3(8, 2, 128), 128, 0, stream>>>(
      anbf2_p, xw3bf_p, h3_p, 128, 256, 128, 16384, 32768, 32768, b3, nullptr);
  k_readout<<<128, 256, 0, stream>>>(h3_p, ws + x3, Bb, 128, 256);

  // ---- head
  k_relu_sum3<<<256, 256, 0, stream>>>(ws + x1, ws + x2, ws + x3, ws + gsum, 65536);
  k_head_gemm<<<128, 256, 0, stream>>>(ws + gsum, l1w, l1b, ws + g1, Bb, 512, 256, 1);
  k_head_gemm<<<64, 256, 0, stream>>>(ws + g1, l2w, l2b, xout, Bb, 256, 128, 1);
  k_head_gemm<<<5, 256, 0, stream>>>(xout, l3w, l3b, ws + lgts, Bb, 128, 10, 0);
  k_logsoftmax<<<1, 128, 0, stream>>>(ws + lgts, logp, Bb, 10);
}
